// INNv2Viz_44976897523930
// MI455X (gfx1250) — matrix operand order repaired and checked by reading, not compiled, not run on hardware
//
#include <hip/hip_runtime.h>
#include <hip/hip_bf16.h>

// ---------------- model constants ----------------
#define VV   32000
#define NN_  8
#define DD   512
#define NL_  2
#define DI_  1024      // 2*D
#define DS_  16
#define DC_  4
#define DR_  32
#define HH   4
#define BB   2
#define LL   1024
#define BN_  16        // BB*NN_
#define TT   16384     // BN_*LL  (mamba tokens)
#define GG   2048      // BB*LL   (attn groups / logits rows)

typedef _Float16 h8   __attribute__((ext_vector_type(8)));
typedef _Float16 v16h __attribute__((ext_vector_type(16)));
typedef float    v8f  __attribute__((ext_vector_type(8)));

// ---------------- CDNA5 async copy helpers (inline asm; ASYNCcnt-tracked) ----------------
__device__ __forceinline__ void async_copy_16B(const _Float16* gsrc, _Float16* ldst)
{
    unsigned long long ga = (unsigned long long)gsrc;
    unsigned int       la = (unsigned int)(unsigned long long)ldst; // generic->LDS offset (addr[31:0])
    asm volatile("global_load_async_to_lds_b128 %0, %1, off"
                 :: "v"(la), "v"(ga) : "memory");
}
__device__ __forceinline__ void wait_async0()
{
    asm volatile("s_wait_asynccnt 0x0" ::: "memory");
}

// ---------------- fragment helpers (contiguous-k 16-bit WMMA operand layout used here) ----------------
__device__ __forceinline__ v16h frag_a_from(const _Float16* row, int g) {
    // A 16x32 f16: lane group g: halves 0..15 -> K=16g..16g+15 contiguous
    h8 lo = *(const h8*)(row + 16 * g);
    h8 hi = *(const h8*)(row + 16 * g + 8);
    return __builtin_shufflevector(lo, hi, 0,1,2,3,4,5,6,7,8,9,10,11,12,13,14,15);
}

// ---------------- wave-level WMMA GEMM (one wave per 32x64 tile) ----------------
// C[M,N] = A[M,K] * W[N,K]^T (+bias) (+=C).  grid=(M/32, N/64), block=32.
template<bool ADD_C, bool HAS_BIAS>
__global__ __launch_bounds__(32)
void gemm_f16_wmma(const _Float16* __restrict__ A, int lda,
                   const _Float16* __restrict__ W, int ldw,
                   float* __restrict__ C, int ldc,
                   const float* __restrict__ bias,
                   int K)
{
    const int lane = threadIdx.x;
    const int g    = lane >> 4;
    const int r    = lane & 15;
    const int m0   = blockIdx.x * 32;
    const int n0   = blockIdx.y * 64;

    const _Float16* aRow0 = A + (size_t)(m0 + r)      * lda;
    const _Float16* aRow1 = A + (size_t)(m0 + 16 + r) * lda;
    const _Float16* wRow0 = W + (size_t)(n0 +  0 + r) * ldw;
    const _Float16* wRow1 = W + (size_t)(n0 + 16 + r) * ldw;
    const _Float16* wRow2 = W + (size_t)(n0 + 32 + r) * ldw;
    const _Float16* wRow3 = W + (size_t)(n0 + 48 + r) * ldw;

    const v8f vz = {0.f,0.f,0.f,0.f,0.f,0.f,0.f,0.f};
    v8f acc[2][4];
    for (int mt = 0; mt < 2; ++mt)
        for (int nt = 0; nt < 4; ++nt)
            acc[mt][nt] = vz;

    for (int k = 0; k < K; k += 32) {
        if (k + 32 < K) {
            __builtin_prefetch(aRow0 + k + 32, 0, 1);   // global_prefetch_b8
            __builtin_prefetch(wRow0 + k + 32, 0, 1);
        }
        v16h a0 = frag_a_from(aRow0 + k, g);
        v16h a1 = frag_a_from(aRow1 + k, g);
        // B 32x16 f16: lane group g holds K = 16g..16g+15 contiguous
        v16h b0 = *(const v16h*)(wRow0 + k + 16 * g);
        v16h b1 = *(const v16h*)(wRow1 + k + 16 * g);
        v16h b2 = *(const v16h*)(wRow2 + k + 16 * g);
        v16h b3 = *(const v16h*)(wRow3 + k + 16 * g);

        acc[0][0] = __builtin_amdgcn_wmma_f32_16x16x32_f16(false, a0, false, b0, (short)0, acc[0][0], false, false);
        acc[0][1] = __builtin_amdgcn_wmma_f32_16x16x32_f16(false, a0, false, b1, (short)0, acc[0][1], false, false);
        acc[0][2] = __builtin_amdgcn_wmma_f32_16x16x32_f16(false, a0, false, b2, (short)0, acc[0][2], false, false);
        acc[0][3] = __builtin_amdgcn_wmma_f32_16x16x32_f16(false, a0, false, b3, (short)0, acc[0][3], false, false);
        acc[1][0] = __builtin_amdgcn_wmma_f32_16x16x32_f16(false, a1, false, b0, (short)0, acc[1][0], false, false);
        acc[1][1] = __builtin_amdgcn_wmma_f32_16x16x32_f16(false, a1, false, b1, (short)0, acc[1][1], false, false);
        acc[1][2] = __builtin_amdgcn_wmma_f32_16x16x32_f16(false, a1, false, b2, (short)0, acc[1][2], false, false);
        acc[1][3] = __builtin_amdgcn_wmma_f32_16x16x32_f16(false, a1, false, b3, (short)0, acc[1][3], false, false);
    }

    for (int mt = 0; mt < 2; ++mt) {
        const int rowBase = m0 + mt * 16 + 8 * g;
        for (int nt = 0; nt < 4; ++nt) {
            const int col = n0 + nt * 16 + r;
            float bv = 0.f;
            if (HAS_BIAS) bv = bias[col];
            float* cp = C + (size_t)rowBase * ldc + col;
            for (int j = 0; j < 8; ++j) {
                float v = acc[mt][nt][j] + bv;
                if (ADD_C) cp[(size_t)j * ldc] += v;
                else       cp[(size_t)j * ldc]  = v;
            }
        }
    }
}

// ---------------- LDS-staged async double-buffered WMMA GEMM ----------------
// 8 waves (4M x 2N), block tile 128x128, each wave a 32x64 register tile.
// A/B k-chunks (128x32 halfs each) staged via GLOBAL_LOAD_ASYNC_TO_LDS_B128,
// double-buffered; fill of step s+1 overlaps WMMAs of step s.
// grid=(M/128, N/128), block=256.  Requires M%128==0, N%128==0, K%32==0.
template<bool ADD_C, bool HAS_BIAS>
__global__ __launch_bounds__(256)
void gemm_f16_wmma_lds(const _Float16* __restrict__ A, int lda,
                       const _Float16* __restrict__ W, int ldw,
                       float* __restrict__ C, int ldc,
                       const float* __restrict__ bias,
                       int K)
{
    __shared__ alignas(128) _Float16 As[2][128][32];
    __shared__ alignas(128) _Float16 Bs[2][128][32];

    const int tid   = threadIdx.x;
    const int wid   = tid >> 5;        // 0..7
    const int lane  = tid & 31;
    const int g     = lane >> 4;
    const int r     = lane & 15;
    const int waveM = wid >> 1;        // 0..3
    const int waveN = wid & 1;         // 0..1

    const int m0 = blockIdx.x * 128;
    const int n0 = blockIdx.y * 128;

    // cooperative fill: thread t copies 32B of one row: row=t>>1, 16-half chunk=(t&1)*16
    const int cpRow = tid >> 1;
    const int cpOff = (tid & 1) * 16;
    const _Float16* gA = A + (size_t)(m0 + cpRow) * lda + cpOff;
    const _Float16* gB = W + (size_t)(n0 + cpRow) * ldw + cpOff;

    auto issue = [&](int buf, int k) {
        const _Float16* sa = gA + k;
        const _Float16* sb = gB + k;
        _Float16* da = &As[buf][cpRow][cpOff];
        _Float16* db = &Bs[buf][cpRow][cpOff];
        async_copy_16B(sa,     da);
        async_copy_16B(sa + 8, da + 8);
        async_copy_16B(sb,     db);
        async_copy_16B(sb + 8, db + 8);
    };

    const v8f vz = {0.f,0.f,0.f,0.f,0.f,0.f,0.f,0.f};
    v8f acc[2][4];
    for (int mt = 0; mt < 2; ++mt)
        for (int nt = 0; nt < 4; ++nt)
            acc[mt][nt] = vz;

    const int mrow = waveM * 32;
    const int nrow = waveN * 64;
    const int nsteps = K >> 5;

    issue(0, 0);
    for (int s = 0; s < nsteps; ++s) {
        const int buf = s & 1;
        wait_async0();            // current buffer's async fills complete (this wave)
        __syncthreads();          // ...and visible across all waves
        if (s + 1 < nsteps) issue(buf ^ 1, (s + 1) * 32);   // overlap next fill with compute

        v16h a0 = frag_a_from(&As[buf][mrow + r][0], g);
        v16h a1 = frag_a_from(&As[buf][mrow + 16 + r][0], g);
        v16h b0 = *(const v16h*)(&Bs[buf][nrow +  0 + r][16 * g]);
        v16h b1 = *(const v16h*)(&Bs[buf][nrow + 16 + r][16 * g]);
        v16h b2 = *(const v16h*)(&Bs[buf][nrow + 32 + r][16 * g]);
        v16h b3 = *(const v16h*)(&Bs[buf][nrow + 48 + r][16 * g]);

        acc[0][0] = __builtin_amdgcn_wmma_f32_16x16x32_f16(false, a0, false, b0, (short)0, acc[0][0], false, false);
        acc[0][1] = __builtin_amdgcn_wmma_f32_16x16x32_f16(false, a0, false, b1, (short)0, acc[0][1], false, false);
        acc[0][2] = __builtin_amdgcn_wmma_f32_16x16x32_f16(false, a0, false, b2, (short)0, acc[0][2], false, false);
        acc[0][3] = __builtin_amdgcn_wmma_f32_16x16x32_f16(false, a0, false, b3, (short)0, acc[0][3], false, false);
        acc[1][0] = __builtin_amdgcn_wmma_f32_16x16x32_f16(false, a1, false, b0, (short)0, acc[1][0], false, false);
        acc[1][1] = __builtin_amdgcn_wmma_f32_16x16x32_f16(false, a1, false, b1, (short)0, acc[1][1], false, false);
        acc[1][2] = __builtin_amdgcn_wmma_f32_16x16x32_f16(false, a1, false, b2, (short)0, acc[1][2], false, false);
        acc[1][3] = __builtin_amdgcn_wmma_f32_16x16x32_f16(false, a1, false, b3, (short)0, acc[1][3], false, false);

        __syncthreads();          // all waves done reading buf before it is refilled
    }

    for (int mt = 0; mt < 2; ++mt) {
        const int rowBase = m0 + mrow + mt * 16 + 8 * g;
        for (int nt = 0; nt < 4; ++nt) {
            const int col = n0 + nrow + nt * 16 + r;
            float bv = 0.f;
            if (HAS_BIAS) bv = bias[col];
            float* cp = C + (size_t)rowBase * ldc + col;
            for (int j = 0; j < 8; ++j) {
                float v = acc[mt][nt][j] + bv;
                if (ADD_C) cp[(size_t)j * ldc] += v;
                else       cp[(size_t)j * ldc]  = v;
            }
        }
    }
}

// ---------------- elementwise / gather kernels ----------------
__global__ void k_embed(const int* __restrict__ x, const float* __restrict__ emb,
                        float* __restrict__ h)
{
    int row = blockIdx.x;            // bn*L + l
    int bn  = row >> 10;
    int l   = row & (LL - 1);
    int b   = bn >> 3;
    int tok = x[b * LL + l];
    const float* e = emb + (size_t)tok * DD;
    float* o = h + (size_t)row * DD;
    for (int d = threadIdx.x; d < DD; d += blockDim.x) o[d] = e[d];
}

__global__ void k_cvt_f16(const float* __restrict__ src, _Float16* __restrict__ dst, size_t n)
{
    size_t i = (size_t)blockIdx.x * blockDim.x + threadIdx.x;
    size_t stride = (size_t)gridDim.x * blockDim.x;
    for (; i < n; i += stride) dst[i] = (_Float16)src[i];
}

__global__ void k_cvt_f16_2d(const float* __restrict__ src, int srcStride,
                             _Float16* __restrict__ dst, int dstStride,
                             size_t rows, int cols)
{
    size_t i = (size_t)blockIdx.x * blockDim.x + threadIdx.x;
    size_t stride = (size_t)gridDim.x * blockDim.x;
    size_t total = rows * (size_t)cols;
    for (; i < total; i += stride) {
        size_t r = i / cols;
        int    c = (int)(i - r * cols);
        dst[r * dstStride + c] = (_Float16)src[r * srcStride + c];
    }
}

__device__ inline float block_reduce_sum256(float v, float* red)
{
    int tid = threadIdx.x;
    red[tid] = v; __syncthreads();
    for (int st = 128; st > 0; st >>= 1) {
        if (tid < st) red[tid] += red[tid + st];
        __syncthreads();
    }
    float r = red[0];
    __syncthreads();
    return r;
}

__global__ __launch_bounds__(256)
void k_layernorm_f16(const float* __restrict__ x, const float* __restrict__ w,
                     const float* __restrict__ b, _Float16* __restrict__ out)
{
    __shared__ float red[256];
    const int row = blockIdx.x;
    const float* xr = x + (size_t)row * DD;
    float s = 0.f;
    for (int d = threadIdx.x; d < DD; d += 256) s += xr[d];
    float mean = block_reduce_sum256(s, red) * (1.f / DD);
    float v = 0.f;
    for (int d = threadIdx.x; d < DD; d += 256) { float t = xr[d] - mean; v += t * t; }
    float var = block_reduce_sum256(v, red) * (1.f / DD);
    float rstd = rsqrtf(var + 1e-5f);
    for (int d = threadIdx.x; d < DD; d += 256)
        out[(size_t)row * DD + d] = (_Float16)((xr[d] - mean) * rstd * w[d] + b[d]);
}

__global__ void k_conv_silu(const float* __restrict__ xz, const float* __restrict__ cw,
                            const float* __restrict__ cb, float* __restrict__ xc,
                            _Float16* __restrict__ xc16)
{
    int row = blockIdx.x;
    int c   = blockIdx.y * 256 + threadIdx.x;
    int l   = row & (LL - 1);
    const float* w = cw + (size_t)c * DC_;
    float acc = cb[c];
    for (int j = 0; j < DC_; ++j) {
        int ll = l - (DC_ - 1) + j;
        if (ll >= 0) acc += w[j] * xz[(size_t)(row - (DC_ - 1) + j) * (2 * DI_) + c];
    }
    float sv = acc / (1.f + __expf(-acc));
    xc  [(size_t)row * DI_ + c] = sv;
    xc16[(size_t)row * DI_ + c] = (_Float16)sv;
}

__global__ void k_softplus(float* __restrict__ v, size_t n)
{
    size_t i = (size_t)blockIdx.x * blockDim.x + threadIdx.x;
    size_t stride = (size_t)gridDim.x * blockDim.x;
    for (; i < n; i += stride) {
        float x = v[i];
        v[i] = (x > 20.f) ? x : log1pf(__expf(x));
    }
}

__global__ __launch_bounds__(256)
void k_scan(const float* __restrict__ dt, const float* __restrict__ xc,
            const float* __restrict__ dbl, const float* __restrict__ A_log,
            float* __restrict__ ys)
{
    __shared__ float BC[2 * DS_];
    const int bn = blockIdx.x;
    const int c  = blockIdx.y * 256 + threadIdx.x;
    float a[DS_], hst[DS_];
    for (int s = 0; s < DS_; ++s) {
        a[s]   = -__expf(A_log[(size_t)c * DS_ + s]);
        hst[s] = 0.f;
    }
    for (int l = 0; l < LL; ++l) {
        const size_t tok = (size_t)bn * LL + l;
        if (threadIdx.x < 2 * DS_) BC[threadIdx.x] = dbl[tok * 64 + DR_ + threadIdx.x];
        __syncthreads();
        float d  = dt[tok * DI_ + c];
        float xv = xc[tok * DI_ + c];
        float dx = d * xv;
        float y = 0.f;
        for (int s = 0; s < DS_; ++s) {
            hst[s] = __expf(d * a[s]) * hst[s] + dx * BC[s];
            y += hst[s] * BC[DS_ + s];
        }
        ys[tok * DI_ + c] = y;
        __syncthreads();
    }
}

__global__ void k_ygate(const float* __restrict__ ys, const float* __restrict__ xc,
                        const float* __restrict__ Dp, const float* __restrict__ xz,
                        _Float16* __restrict__ y16)
{
    int row = blockIdx.x;
    int c   = blockIdx.y * 256 + threadIdx.x;
    float z = xz[(size_t)row * (2 * DI_) + DI_ + c];
    float y = (ys[(size_t)row * DI_ + c] + Dp[c] * xc[(size_t)row * DI_ + c])
            * (z / (1.f + __expf(-z)));
    y16[(size_t)row * DI_ + c] = (_Float16)y;
}

__global__ __launch_bounds__(128)
void k_attn(const float* __restrict__ qkv, _Float16* __restrict__ o16)
{
    __shared__ float Qs[NN_][128], Ks[NN_][128], Vs[NN_][128];
    __shared__ float sc[NN_][NN_];
    const int grp  = blockIdx.x;
    const int head = blockIdx.y;
    const int b = grp >> 10;
    const int l = grp & (LL - 1);
    const int d = threadIdx.x;
    for (int n = 0; n < NN_; ++n) {
        size_t rq = ((size_t)(b * NN_ + n) * LL + l) * (3 * DD);
        Qs[n][d] = qkv[rq +          head * 128 + d];
        Ks[n][d] = qkv[rq + DD     + head * 128 + d];
        Vs[n][d] = qkv[rq + 2 * DD + head * 128 + d];
    }
    __syncthreads();
    if (threadIdx.x < NN_ * NN_) {
        int n = threadIdx.x >> 3, m = threadIdx.x & 7;
        float s = 0.f;
        for (int k = 0; k < 128; ++k) s += Qs[n][k] * Ks[m][k];
        sc[n][m] = s * 0.08838834764831845f;
    }
    __syncthreads();
    if (threadIdx.x < NN_) {
        int n = threadIdx.x;
        float mx = sc[n][0];
        for (int m = 1; m < NN_; ++m) mx = fmaxf(mx, sc[n][m]);
        float e[NN_]; float sum = 0.f;
        for (int m = 0; m < NN_; ++m) { e[m] = __expf(sc[n][m] - mx); sum += e[m]; }
        float inv = 1.f / sum;
        for (int m = 0; m < NN_; ++m) sc[n][m] = e[m] * inv;
    }
    __syncthreads();
    for (int n = 0; n < NN_; ++n) {
        float o = 0.f;
        for (int m = 0; m < NN_; ++m) o += sc[n][m] * Vs[m][d];
        o16[((size_t)(b * NN_ + n) * LL + l) * DD + head * 128 + d] = (_Float16)o;
    }
}

__global__ void k_mean(const float* __restrict__ h, float* __restrict__ hm)
{
    int g = blockIdx.x;
    int b = g >> 10, l = g & (LL - 1);
    for (int d = threadIdx.x; d < DD; d += 128) {
        float s = 0.f;
        for (int n = 0; n < NN_; ++n)
            s += h[((size_t)(b * NN_ + n) * LL + l) * DD + d];
        hm[(size_t)g * DD + d] = s * (1.f / NN_);
    }
}

// ---------------- host orchestration ----------------
extern "C" void kernel_launch(void* const* d_in, const int* in_sizes, int n_in,
                              void* d_out, int out_size, void* d_ws, size_t ws_size,
                              hipStream_t stream)
{
    (void)in_sizes; (void)n_in; (void)out_size; (void)ws_size;
    const int*   x          = (const int*)  d_in[0];
    const float* emb        = (const float*)d_in[1];
    const float* in_proj_w  = (const float*)d_in[2];
    const float* conv_w     = (const float*)d_in[3];
    const float* conv_b     = (const float*)d_in[4];
    const float* x_proj_w   = (const float*)d_in[5];
    const float* dt_proj_w  = (const float*)d_in[6];
    const float* dt_proj_b  = (const float*)d_in[7];
    const float* A_log      = (const float*)d_in[8];
    const float* D_param    = (const float*)d_in[9];
    const float* out_proj_w = (const float*)d_in[10];
    const float* attn_in_w  = (const float*)d_in[11];
    const float* attn_in_b  = (const float*)d_in[12];
    const float* attn_out_w = (const float*)d_in[13];
    const float* attn_out_b = (const float*)d_in[14];
    const float* norm1_w    = (const float*)d_in[15];
    const float* norm1_b    = (const float*)d_in[16];
    const float* norm2_w    = (const float*)d_in[17];
    const float* norm2_b    = (const float*)d_in[18];
    const float* normf_w    = (const float*)d_in[19];
    const float* normf_b    = (const float*)d_in[20];
    const float* head_b     = (const float*)d_in[21];
    float* logits = (float*)d_out;

    char* p = (char*)d_ws;
    auto carve = [&](size_t bytes) -> char* {
        char* q = p; p += (bytes + 255) & ~(size_t)255; return q;
    };
    float*    hbuf  = (float*)carve((size_t)TT * DD * 4);
    float*    xz    = (float*)carve((size_t)TT * 2 * DI_ * 4);
    float*    xc    = (float*)carve((size_t)TT * DI_ * 4);
    float*    dbl   = (float*)carve((size_t)TT * 64 * 4);
    float*    dt    = (float*)carve((size_t)TT * DI_ * 4);
    float*    ys    = (float*)carve((size_t)TT * DI_ * 4);
    float*    hm    = (float*)carve((size_t)GG * DD * 4);
    _Float16* emb16 = (_Float16*)carve((size_t)VV * DD * 2);
    _Float16* xn16  = (_Float16*)carve((size_t)TT * DD * 2);
    _Float16* xc16  = (_Float16*)carve((size_t)TT * DI_ * 2);
    _Float16* y16   = (_Float16*)carve((size_t)TT * DI_ * 2);
    _Float16* ao16  = (_Float16*)carve((size_t)TT * DD * 2);
    _Float16* dt16  = (_Float16*)carve((size_t)TT * DR_ * 2);
    _Float16* w_in16  = (_Float16*)carve((size_t)2 * DI_ * DD * 2);
    _Float16* w_xp16  = (_Float16*)carve((size_t)(DR_ + 2 * DS_) * DI_ * 2);
    _Float16* w_dtp16 = (_Float16*)carve((size_t)DI_ * DR_ * 2);
    _Float16* w_op16  = (_Float16*)carve((size_t)DD * DI_ * 2);
    _Float16* w_ai16  = (_Float16*)carve((size_t)3 * DD * DD * 2);
    _Float16* w_ao16  = (_Float16*)carve((size_t)DD * DD * 2);
    float* qkv = xz;   // alias: xz dead after z-gate; qkv (TT*1536) fits in TT*2048

    k_embed<<<TT, 128, 0, stream>>>(x, emb, hbuf);
    k_cvt_f16<<<2048, 256, 0, stream>>>(emb, emb16, (size_t)VV * DD);

    for (int i = 0; i < NL_; ++i) {
        k_cvt_f16<<<1024, 256, 0, stream>>>(in_proj_w  + (size_t)i * 2 * DI_ * DD, w_in16,  (size_t)2 * DI_ * DD);
        k_cvt_f16<<<256 , 256, 0, stream>>>(x_proj_w   + (size_t)i * 64 * DI_,     w_xp16,  (size_t)64 * DI_);
        k_cvt_f16<<<128 , 256, 0, stream>>>(dt_proj_w  + (size_t)i * DI_ * DR_,    w_dtp16, (size_t)DI_ * DR_);
        k_cvt_f16<<<512 , 256, 0, stream>>>(out_proj_w + (size_t)i * DD * DI_,     w_op16,  (size_t)DD * DI_);
        k_cvt_f16<<<768 , 256, 0, stream>>>(attn_in_w  + (size_t)i * 3 * DD * DD,  w_ai16,  (size_t)3 * DD * DD);
        k_cvt_f16<<<256 , 256, 0, stream>>>(attn_out_w + (size_t)i * DD * DD,      w_ao16,  (size_t)DD * DD);

        // ---- Mamba branch ----
        k_layernorm_f16<<<TT, 256, 0, stream>>>(hbuf, norm1_w + i * DD, norm1_b + i * DD, xn16);
        // xz = xn @ in_proj_w^T   (16384 x 2048, K=512)
        gemm_f16_wmma_lds<false, false><<<dim3(TT / 128, 2 * DI_ / 128), 256, 0, stream>>>(
            xn16, DD, w_in16, DD, xz, 2 * DI_, nullptr, DD);
        k_conv_silu<<<dim3(TT, DI_ / 256), 256, 0, stream>>>(
            xz, conv_w + (size_t)i * DI_ * DC_, conv_b + i * DI_, xc, xc16);
        // dbl = xc @ x_proj_w^T   (16384 x 64, K=1024)  -- N=64: wave-level kernel
        gemm_f16_wmma<false, false><<<dim3(TT / 32, 64 / 64), 32, 0, stream>>>(
            xc16, DI_, w_xp16, DI_, dbl, 64, nullptr, DI_);
        k_cvt_f16_2d<<<1024, 256, 0, stream>>>(dbl, 64, dt16, DR_, (size_t)TT, DR_);
        // dt = dt16 @ dt_proj_w^T + dt_proj_b  (16384 x 1024, K=32)
        gemm_f16_wmma_lds<false, true><<<dim3(TT / 128, DI_ / 128), 256, 0, stream>>>(
            dt16, DR_, w_dtp16, DR_, dt, DI_, dt_proj_b + i * DI_, DR_);
        k_softplus<<<2048, 256, 0, stream>>>(dt, (size_t)TT * DI_);
        k_scan<<<dim3(BN_, DI_ / 256), 256, 0, stream>>>(
            dt, xc, dbl, A_log + (size_t)i * DI_ * DS_, ys);
        k_ygate<<<dim3(TT, DI_ / 256), 256, 0, stream>>>(
            ys, xc, D_param + i * DI_, xz, y16);
        // h += y @ out_proj_w^T   (16384 x 512, K=1024)
        gemm_f16_wmma_lds<true, false><<<dim3(TT / 128, DD / 128), 256, 0, stream>>>(
            y16, DI_, w_op16, DI_, hbuf, DD, nullptr, DI_);

        // ---- attention branch ----
        k_layernorm_f16<<<TT, 256, 0, stream>>>(hbuf, norm2_w + i * DD, norm2_b + i * DD, xn16);
        // qkv = xn2 @ attn_in_w^T + b   (16384 x 1536, K=512)  [qkv aliases xz]
        gemm_f16_wmma_lds<false, true><<<dim3(TT / 128, 3 * DD / 128), 256, 0, stream>>>(
            xn16, DD, w_ai16, DD, qkv, 3 * DD, attn_in_b + i * 3 * DD, DD);
        k_attn<<<dim3(GG, HH), 128, 0, stream>>>(qkv, ao16);
        // h += attnO @ attn_out_w^T + b  (16384 x 512, K=512)
        gemm_f16_wmma_lds<true, true><<<dim3(TT / 128, DD / 128), 256, 0, stream>>>(
            ao16, DD, w_ao16, DD, hbuf, DD, attn_out_b + i * DD, DD);
    }

    k_mean<<<GG, 128, 0, stream>>>(hbuf, hm);
    k_layernorm_f16<<<GG, 256, 0, stream>>>(hm, normf_w, normf_b, xn16);
    // logits = hf @ embedding^T + head_b   (2048 x 32000, K=512)
    gemm_f16_wmma_lds<false, true><<<dim3(GG / 128, VV / 128), 256, 0, stream>>>(
        xn16, DD, emb16, DD, logits, VV, head_b, DD);
}
